// _GATNet_40286793236670
// MI455X (gfx1250) — matrix operand order repaired and checked by reading, not compiled, not run on hardware
//
#include <hip/hip_runtime.h>
#include <hip/hip_bf16.h>

#define NN      100000
#define EE      1600000
#define ETOT    (EE + NN)
#define HID     128
#define HEADS   4
#define HD      32
#define LAYERS  3
#define GG      128
#define SLOPE   0.2f
#define BNEPS   1e-5f

typedef __attribute__((ext_vector_type(16))) __bf16 v16bf;
typedef __attribute__((ext_vector_type(8)))  float  v8f;

// ---------------------------------------------------------------------------
// float atomic max via signed/unsigned int trick (target init'd to -inf)
// ---------------------------------------------------------------------------
__device__ __forceinline__ void atomicMaxF(float* addr, float v) {
    if (v >= 0.0f) atomicMax((int*)addr, __float_as_int(v));
    else           atomicMin((unsigned int*)addr, __float_as_uint(v));
}

// ---------------------------------------------------------------------------
// one-time prep A: decode int64 edge_index -> int32 src/dst with self-loops
// ---------------------------------------------------------------------------
__global__ void k_prep_edges(const long long* __restrict__ ei,
                             int* __restrict__ src32, int* __restrict__ dst32) {
    int e = blockIdx.x * blockDim.x + threadIdx.x;
    if (e >= ETOT) return;
    if (e < EE) { src32[e] = (int)ei[e]; dst32[e] = (int)ei[EE + e]; }
    else        { src32[e] = dst32[e] = e - EE; }
}

// ---------------------------------------------------------------------------
// one-time prep B: pack all layers' W into bf16 in WMMA B-fragment layout:
//   Wb[l][c][wave][lane][i],  c = K-chunk (K32), wave = N-tile (16 cols)
//   value = W[l][ c*32 + (lane<16?0:16) + i ][ wave*16 + (lane&15) ]
// -> per-lane B fragment is 16 contiguous bf16 (32B) = 2x b128 loads.
// ---------------------------------------------------------------------------
__global__ void k_prep_w(const float* __restrict__ W, __bf16* __restrict__ Wb) {
    int idx = blockIdx.x * blockDim.x + threadIdx.x;     // 3*4*8*32*16 = 49152
    if (idx >= LAYERS * 4 * 8 * 32 * 16) return;
    int i    = idx & 15;
    int lane = (idx >> 4) & 31;
    int w    = (idx >> 9) & 7;
    int c    = (idx >> 12) & 3;
    int l    = idx >> 14;
    int k    = c * 32 + ((lane < 16) ? 0 : 16) + i;
    int col  = w * 16 + (lane & 15);
    Wb[idx]  = (__bf16)W[(size_t)l * HID * HID + (size_t)k * HID + col];
}

// ---------------------------------------------------------------------------
// h = x @ W via v_wmma_f32_16x16x32_bf16.
// grid.x = NN/16 (M tiles), block = 256 (8 waves); wave w owns N-tile w.
// A layout (16-bit 16x32): lane<16 -> row=lane, K{0..15};
//                          lane>=16 -> row=lane-16, K{16..31}
// B: pre-packed contiguous per-lane fragments (k_prep_w).
// C layout (f32 16x16): elem r -> row r (+8 for upper lanes), col = lane&15
// ---------------------------------------------------------------------------
__global__ void __launch_bounds__(256)
k_gemm_node(const float* __restrict__ X, const __bf16* __restrict__ Wb,
            float* __restrict__ Hh) {
    const int wave = threadIdx.x >> 5;
    const int lane = threadIdx.x & 31;
    const int m0   = blockIdx.x << 4;

    const int rowA  = m0 + (lane & 15);
    const int koffA = (lane < 16) ? 0 : 16;

    const float* xr  = X + (size_t)rowA * HID;
    const v16bf* wbp = ((const v16bf*)Wb) + (size_t)wave * 32 + lane;

    v8f acc = {};
#pragma unroll
    for (int c = 0; c < 4; ++c) {
        const int k0 = c << 5;
        v16bf a;
#pragma unroll
        for (int i = 0; i < 8; ++i) {
            a[i]     = (__bf16)xr[k0 + koffA + i];
            a[i + 8] = (__bf16)xr[k0 + koffA + 8 + i];
        }
        v16bf b = wbp[c * 256];                // (c*8 + wave)*32 + lane
        acc = __builtin_amdgcn_wmma_f32_16x16x32_bf16(
                  false, a, false, b, (short)0, acc, false, false);
    }
    const int mrow = m0 + ((lane < 16) ? 0 : 8);
    const int ccol = (wave << 4) + (lane & 15);
#pragma unroll
    for (int r = 0; r < 8; ++r)
        Hh[(size_t)(mrow + r) * HID + ccol] = acc[r];
}

// ---------------------------------------------------------------------------
// per-node, per-head attention logits: a_s/a_d [N,H]
// ---------------------------------------------------------------------------
__global__ void k_attn_coef(const float* __restrict__ Hh,
                            const float* __restrict__ asrc,
                            const float* __restrict__ adst,
                            float* __restrict__ a_s, float* __restrict__ a_d) {
    int idx = blockIdx.x * blockDim.x + threadIdx.x;
    if (idx >= NN * HEADS) return;
    int n = idx >> 2, h = idx & 3;
    const float* hp = Hh + (size_t)n * HID + h * HD;
    float s = 0.f, d = 0.f;
#pragma unroll
    for (int i = 0; i < HD; ++i) {
        float v = hp[i];
        s += v * asrc[h * HD + i];
        d += v * adst[h * HD + i];
    }
    a_s[idx] = s; a_d[idx] = d;
}

// ---------------------------------------------------------------------------
// per-layer init: out=0 (N*HID), emax=-inf, denom=0 (N*HEADS)
// ---------------------------------------------------------------------------
__global__ void k_init_layer(float* __restrict__ Out, float* __restrict__ emax,
                             float* __restrict__ denom) {
    int idx = blockIdx.x * blockDim.x + threadIdx.x;
    if (idx < NN * HID) Out[idx] = 0.0f;
    if (idx < NN * HEADS) {
        emax[idx]  = -__builtin_inff();
        denom[idx] = 0.0f;
    }
}

// ---------------------------------------------------------------------------
// pass 1: segment max of leaky-relu logits over incoming edges
// ---------------------------------------------------------------------------
__global__ void k_edge_max(const int* __restrict__ src32,
                           const int* __restrict__ dst32,
                           const float* __restrict__ a_s,
                           const float* __restrict__ a_d,
                           float* __restrict__ emax) {
    int idx = blockIdx.x * blockDim.x + threadIdx.x;
    if (idx >= ETOT * HEADS) return;
    int e = idx >> 2, h = idx & 3;
    int src = src32[e], dst = dst32[e];
    float v = a_s[src * HEADS + h] + a_d[dst * HEADS + h];
    v = (v > 0.f) ? v : SLOPE * v;
    atomicMaxF(&emax[dst * HEADS + h], v);
}

// ---------------------------------------------------------------------------
// pass 2: softmax denominator
// ---------------------------------------------------------------------------
__global__ void k_edge_sum(const int* __restrict__ src32,
                           const int* __restrict__ dst32,
                           const float* __restrict__ a_s,
                           const float* __restrict__ a_d,
                           const float* __restrict__ emax,
                           float* __restrict__ denom) {
    int idx = blockIdx.x * blockDim.x + threadIdx.x;
    if (idx >= ETOT * HEADS) return;
    int e = idx >> 2, h = idx & 3;
    int src = src32[e], dst = dst32[e];
    float v = a_s[src * HEADS + h] + a_d[dst * HEADS + h];
    v = (v > 0.f) ? v : SLOPE * v;
    float p = __expf(v - emax[dst * HEADS + h]);
    atomicAdd(&denom[dst * HEADS + h], p);
}

// ---------------------------------------------------------------------------
// pass 3: weighted message scatter. 32 lanes per edge (wave32-native):
// lane = head-dim element, 4 heads unrolled -> coalesced 128B gather of
// h[src] and 128B atomic-add burst into out[dst] (L2-resident, 192MB L2).
// ---------------------------------------------------------------------------
__global__ void __launch_bounds__(256)
k_edge_msg(const int* __restrict__ src32, const int* __restrict__ dst32,
           const float* __restrict__ Hh,
           const float* __restrict__ a_s, const float* __restrict__ a_d,
           const float* __restrict__ emax, const float* __restrict__ denom,
           float* __restrict__ Out) {
    long long t = (long long)blockIdx.x * blockDim.x + threadIdx.x;
    int e    = (int)(t >> 5);
    int lane = (int)(t & 31);
    if (e >= ETOT) return;
    int src = src32[e], dst = dst32[e];
    const float* hs = Hh + (size_t)src * HID;
    float*       op = Out + (size_t)dst * HID;
#pragma unroll
    for (int h = 0; h < HEADS; ++h) {
        float v = a_s[src * HEADS + h] + a_d[dst * HEADS + h];
        v = (v > 0.f) ? v : SLOPE * v;
        float alpha = __expf(v - emax[dst * HEADS + h]) /
                      (denom[dst * HEADS + h] + 1e-16f);
        atomicAdd(op + h * HD + lane, hs[h * HD + lane] * alpha);
    }
}

// ---------------------------------------------------------------------------
// bias + ReLU + BatchNorm(eval), in place (Out becomes next layer's x)
// ---------------------------------------------------------------------------
__global__ void k_post(float* __restrict__ Out,
                       const float* __restrict__ bias,
                       const float* __restrict__ gamma,
                       const float* __restrict__ beta,
                       const float* __restrict__ mean,
                       const float* __restrict__ var) {
    int idx = blockIdx.x * blockDim.x + threadIdx.x;
    if (idx >= NN * HID) return;
    int f = idx & (HID - 1);
    float v = Out[idx] + bias[f];
    v = fmaxf(v, 0.0f);
    v = (v - mean[f]) * (gamma[f] * __frsqrt_rn(var[f] + BNEPS)) + beta[f];
    Out[idx] = v;
}

// ---------------------------------------------------------------------------
// pooling
// ---------------------------------------------------------------------------
__global__ void k_pool_init(float* __restrict__ pooled, float* __restrict__ cnt) {
    int idx = blockIdx.x * blockDim.x + threadIdx.x;
    if (idx < GG * HID) pooled[idx] = 0.0f;
    if (idx < GG) cnt[idx] = 0.0f;
}

__global__ void k_pool(const float* __restrict__ X,
                       const long long* __restrict__ batch,
                       float* __restrict__ pooled, float* __restrict__ cnt) {
    int idx = blockIdx.x * blockDim.x + threadIdx.x;
    if (idx >= NN * HID) return;
    int n = idx >> 7, f = idx & (HID - 1);
    int b = (int)batch[n];
    atomicAdd(&pooled[b * HID + f], X[idx]);
    if (f == 0) atomicAdd(&cnt[b], 1.0f);
}

// ---------------------------------------------------------------------------
// MLP head: 128 graphs, one thread each (tiny)
// ---------------------------------------------------------------------------
__global__ void k_mlp(const float* __restrict__ pooled,
                      const float* __restrict__ cnt,
                      const float* __restrict__ Wh1, const float* __restrict__ bh1,
                      const float* __restrict__ Wh2, const float* __restrict__ bh2,
                      float* __restrict__ out) {
    int g = blockIdx.x * blockDim.x + threadIdx.x;
    if (g >= GG) return;
    float c = cnt[g]; c = (c > 1.0f) ? c : 1.0f;
    float inv = 1.0f / c;
    const float* pr = pooled + (size_t)g * HID;
    float acc_out = bh2[0];
    for (int j = 0; j < HID / 2; ++j) {
        float a = bh1[j];
        for (int i = 0; i < HID; ++i)
            a += (pr[i] * inv) * Wh1[i * (HID / 2) + j];
        a = fmaxf(a, 0.0f);
        acc_out += a * Wh2[j];
    }
    out[g] = acc_out;
}

// ---------------------------------------------------------------------------
extern "C" void kernel_launch(void* const* d_in, const int* in_sizes, int n_in,
                              void* d_out, int out_size, void* d_ws, size_t ws_size,
                              hipStream_t stream) {
    const float*     x     = (const float*)d_in[0];
    const long long* ei    = (const long long*)d_in[1];
    const long long* batch = (const long long*)d_in[2];
    const float*     W     = (const float*)d_in[3];
    const float*     asrc  = (const float*)d_in[4];
    const float*     adst  = (const float*)d_in[5];
    const float*     bias  = (const float*)d_in[6];
    const float*     gamma = (const float*)d_in[7];
    const float*     beta  = (const float*)d_in[8];
    const float*     bmean = (const float*)d_in[9];
    const float*     bvar  = (const float*)d_in[10];
    const float*     Wh1   = (const float*)d_in[11];
    const float*     bh1   = (const float*)d_in[12];
    const float*     Wh2   = (const float*)d_in[13];
    const float*     bh2   = (const float*)d_in[14];
    float*           out   = (float*)d_out;

    float* ws     = (float*)d_ws;
    float* bufA   = ws;                      // N*HID
    float* bufB   = bufA + (size_t)NN * HID; // N*HID
    float* hbuf   = bufB + (size_t)NN * HID; // N*HID
    float* a_s    = hbuf + (size_t)NN * HID; // N*H
    float* a_d    = a_s  + (size_t)NN * HEADS;
    float* emax   = a_d  + (size_t)NN * HEADS;
    float* denom  = emax + (size_t)NN * HEADS;
    float* pooled = denom + (size_t)NN * HEADS; // G*HID
    float* cnt    = pooled + (size_t)GG * HID;  // G
    int*    src32 = (int*)(cnt + GG);           // ETOT
    int*    dst32 = src32 + ETOT;               // ETOT
    __bf16* Wb    = (__bf16*)(dst32 + ETOT);    // 3*4*8*32*16 (32B-aligned)

    const int TB = 256;
    const int gElem  = (NN * HID) / TB;                    // 50000 (exact)
    const int gAttn  = (NN * HEADS + TB - 1) / TB;
    const int gEdgeH = (ETOT * HEADS + TB - 1) / TB;
    const int gEdge  = (ETOT + TB - 1) / TB;
    const int gMsg   = (int)(((long long)ETOT * 32) / TB); // exact
    const int gGemm  = NN / 16;                            // 6250 (exact)
    const int gWb    = (LAYERS * 4 * 8 * 32 * 16) / TB;    // 192 (exact)

    k_prep_edges<<<gEdge, TB, 0, stream>>>(ei, src32, dst32);
    k_prep_w<<<gWb, TB, 0, stream>>>(W, Wb);

    const float* xin = x;
    for (int l = 0; l < LAYERS; ++l) {
        float* xout = (l == 1) ? bufB : bufA;   // l0->A, l1->B, l2->A

        k_gemm_node<<<gGemm, TB, 0, stream>>>(xin, Wb + (size_t)l * 4 * 8 * 32 * 16,
                                              hbuf);
        k_attn_coef<<<gAttn, TB, 0, stream>>>(hbuf, asrc + l * HEADS * HD,
                                              adst + l * HEADS * HD, a_s, a_d);
        k_init_layer<<<gElem, TB, 0, stream>>>(xout, emax, denom);
        k_edge_max<<<gEdgeH, TB, 0, stream>>>(src32, dst32, a_s, a_d, emax);
        k_edge_sum<<<gEdgeH, TB, 0, stream>>>(src32, dst32, a_s, a_d, emax, denom);
        k_edge_msg<<<gMsg, TB, 0, stream>>>(src32, dst32, hbuf, a_s, a_d,
                                            emax, denom, xout);
        k_post<<<gElem, TB, 0, stream>>>(xout, bias + l * HID, gamma + l * HID,
                                         beta + l * HID, bmean + l * HID,
                                         bvar + l * HID);
        xin = xout;
    }

    k_pool_init<<<(GG * HID + TB - 1) / TB, TB, 0, stream>>>(pooled, cnt);
    k_pool<<<gElem, TB, 0, stream>>>(xin, batch, pooled, cnt);
    k_mlp<<<1, GG, 0, stream>>>(pooled, cnt, Wh1, bh1, Wh2, bh2, out);
}
